// HyperConvRes_81870666597035
// MI455X (gfx1250) — matrix operand order repaired and checked by reading, not compiled, not run on hardware
//
#include <hip/hip_runtime.h>

#define B_    16
#define CIN   64
#define COUT  64
#define ZDIM_ 64
#define KSZ_  3
#define WH_   32
#define T_    32768
#define K_    128
#define S_    256            // samples per segment (T/K)
#define WROWS 4096           // COUT*CIN, stored out-major
#define XRP   260            // f32 pitch of raw-x LDS tile (16B aligned rows)

typedef __attribute__((ext_vector_type(16))) __bf16 v16bf;
typedef __attribute__((ext_vector_type(8)))  float  v8f;
typedef int v4i_ __attribute__((vector_size(16)));

union V16U8 { v16bf v; unsigned int u[8]; };

#if defined(__gfx1250__) &&                                        \
    __has_builtin(__builtin_amdgcn_global_load_async_to_lds_b128) && \
    __has_builtin(__builtin_amdgcn_s_wait_asynccnt)
#define HAS_ASYNC_LDS 1
#else
#define HAS_ASYNC_LDS 0
#endif

__device__ __forceinline__ unsigned short f2bf(float f) {
  unsigned int u = __float_as_uint(f);
  u += 0x7FFFu + ((u >> 16) & 1u);          // round-to-nearest-even
  return (unsigned short)(u >> 16);
}

// A operand (16xK tile, bf16, row-major [m][k] source, pitch in elements).
// Lane l: row m = mBase + (l&15), half = l>>4.
// Slot pair p (K even): K = kk + (p<4 ? 2p + 8*half : 2(p-4) + 16 + 8*half).
__device__ __forceinline__ v16bf gatherA(const unsigned short* base, int pitch,
                                         int mBase, int kk) {
  int lane = threadIdx.x & 31;
  int hlf  = lane >> 4;
  const unsigned short* row = base + (size_t)(mBase + (lane & 15)) * pitch;
  V16U8 r;
#pragma unroll
  for (int p = 0; p < 8; ++p) {
    int K = kk + ((p < 4) ? (2 * p + 8 * hlf) : (2 * (p - 4) + 16 + 8 * hlf));
    r.u[p] = *(const unsigned int*)(row + K);
  }
  return r.v;
}

// B operand (Kx16 tile, bf16) from a source stored transposed: [n][k] row-major.
// Lane l: col n = nBase + (l&15); slot pair p: K = kk + (p<4 ? 2p + 8*half : 2(p-4) + 16 + 8*half).
__device__ __forceinline__ v16bf gatherB(const unsigned short* base, int pitch,
                                         int nBase, int kk) {
  int lane = threadIdx.x & 31;
  int hlf  = lane >> 4;
  const unsigned short* row = base + (size_t)(nBase + (lane & 15)) * pitch;
  V16U8 r;
#pragma unroll
  for (int p = 0; p < 8; ++p) {
    int K = kk + ((p < 4) ? (2 * p + 8 * hlf) : (2 * (p - 4) + 16 + 8 * hlf));
    r.u[p] = *(const unsigned int*)(row + K);
  }
  return r.v;
}

// ---------------------------------------------------------------------------
// Kernel 1: tiny prep. Blocks [0,256): hypernet hidden layers + bias vectors
// (one (b,k) pair per wave). Blocks [256,320): grid-stride conversions:
//   w2e[out*64+cin][h] = sum_ksz wm2[(ksz*64+cin)*64+out][h]   (bf16)
//   wb2e[out*64+cin]   = sum_ksz wb2[...]                       (f32)
//   rwb/swb            = bf16(rw/sw) row-major [o][c]
// ---------------------------------------------------------------------------
__global__ __launch_bounds__(256)
void prep_kernel(const float* __restrict__ z,
                 const float* __restrict__ wm1, const float* __restrict__ wb1,
                 const float* __restrict__ wm2, const float* __restrict__ wb2,
                 const float* __restrict__ bm1, const float* __restrict__ bb1,
                 const float* __restrict__ bm2, const float* __restrict__ bb2,
                 const float* __restrict__ rw,  const float* __restrict__ sw,
                 unsigned short* __restrict__ hT,
                 unsigned short* __restrict__ w2e,
                 float* __restrict__ wb2e,
                 unsigned short* __restrict__ rwb,
                 unsigned short* __restrict__ swb,
                 float* __restrict__ biasW) {
  __shared__ float hbuf[8][WH_];
  const int tid = threadIdx.x;
  if (blockIdx.x < 256) {
    const int wave = tid >> 5, lane = tid & 31;
    const int p = blockIdx.x * 8 + wave;          // (b,k) pair id
    const int b = p >> 7, k = p & 127;
    const float* zc = z + (size_t)b * ZDIM_ * K_ + k;
    float acc1 = wb1[lane];                       // weight-model hidden unit
    float acc2 = bb1[lane];                       // bias-model hidden unit
    for (int c = 0; c < ZDIM_; ++c) {
      float zv = zc[(size_t)c * K_];
      acc1 += wm1[lane * ZDIM_ + c] * zv;
      acc2 += bm1[lane * ZDIM_ + c] * zv;
    }
    acc1 = fmaxf(acc1, 0.f);
    acc2 = fmaxf(acc2, 0.f);
    hT[(size_t)p * WH_ + lane] = f2bf(acc1);      // hT[b][k][h] (A layout src)
    hbuf[wave][lane] = acc2;
    __syncthreads();
    for (int oo = 0; oo < 2; ++oo) {
      int o = lane + 32 * oo;
      float a = bb2[o];
      for (int h = 0; h < WH_; ++h) a += bm2[o * WH_ + h] * hbuf[wave][h];
      biasW[(size_t)p * COUT + o] = a;            // bias[b][k][out]
    }
  } else {
    const int nb = (int)gridDim.x - 256;
    int gtid = ((int)blockIdx.x - 256) * 256 + tid;
    int stride = nb * 256;
    const int N1 = WROWS * WH_;                   // 131072
    for (int i = gtid; i < N1 + 3 * WROWS; i += stride) {
      if (i < N1) {
        int row = i / WH_, h = i % WH_;           // row = out*64 + cin
        int out = row >> 6, cin = row & 63;
        float s = 0.f;
        for (int ks = 0; ks < KSZ_; ++ks)
          s += wm2[(size_t)((ks * CIN + cin) * COUT + out) * WH_ + h];
        w2e[i] = f2bf(s);
      } else if (i < N1 + WROWS) {
        int row = i - N1;
        int out = row >> 6, cin = row & 63;
        float s = 0.f;
        for (int ks = 0; ks < KSZ_; ++ks)
          s += wb2[(ks * CIN + cin) * COUT + out];
        wb2e[row] = s;
      } else if (i < N1 + 2 * WROWS) {
        int j = i - N1 - WROWS;
        rwb[j] = f2bf(rw[j]);
      } else {
        int j = i - N1 - 2 * WROWS;
        swb[j] = f2bf(sw[j]);
      }
    }
  }
}

// ---------------------------------------------------------------------------
// Kernel 2: hyper-weight generation via WMMA.
// Weff[b][kseg][out*64+cin] = (h^T @ w2e^T)[kseg][out*64+cin] + wb2e
// M = kseg (128 -> 8 tiles), N = 4096 (256 tiles), Kdim = WH = 32 (1 wmma).
// ---------------------------------------------------------------------------
__global__ __launch_bounds__(256)
void wgen_kernel(const unsigned short* __restrict__ hT,
                 const unsigned short* __restrict__ w2e,
                 const float* __restrict__ wb2e,
                 unsigned short* __restrict__ weff) {
  const int wave = threadIdx.x >> 5;
  const int blk = blockIdx.x;                     // 0..4095
  const int b  = blk >> 8;
  const int mt = (blk >> 5) & 7;
  const int nt = (blk & 31) * 8 + wave;           // 0..255
  const int lane = threadIdx.x & 31;
  const int hlf = lane >> 4;

  v16bf a  = gatherA(hT + (size_t)b * K_ * WH_, WH_, mt * 16, 0);
  v16bf bm = gatherB(w2e, WH_, nt * 16, 0);
  v8f c = {};
  c = __builtin_amdgcn_wmma_f32_16x16x32_bf16(false, a, false, bm, (short)0, c,
                                              false, false);
  const int n = nt * 16 + (lane & 15);
  const float wb = wb2e[n];
  unsigned short* dst = weff + (size_t)b * K_ * WROWS + n;
#pragma unroll
  for (int r = 0; r < 8; ++r) {
    int kseg = mt * 16 + r + 8 * hlf;
    dst[(size_t)kseg * WROWS] = f2bf(c[r] + wb);
  }
}

// ---------------------------------------------------------------------------
// Kernel 3: fused main. One block per (b, segment): 256 threads / 8 waves.
//  stage x:  async global->LDS B128 copies (ASYNCcnt) of raw f32 x tile,
//            then one LDS->LDS bf16 transpose-convert pass.
//  stage1: Y = sin(X(256x64) @ Weff(64x64) + bias)      (WMMA bf16)
//  stage2: R = Y @ rw^T + rb ; out1 = (R + x)/2          (WMMA bf16)
//          S = Y @ sw^T + sb ; out2 = S
// Dynamic LDS regions (fixed offsets for both async/sync paths):
//   [0, 66560)       Xraw f32 [64][260]   -- later aliased by Out f32 [64][257]
//   [66560, 103424)  Xa bf16 [256][72]
//   [103424, 140288) Y  bf16 [256][72]
// ---------------------------------------------------------------------------
__global__ __launch_bounds__(256)
void hyperconv_main(const float* __restrict__ x,
                    const unsigned short* __restrict__ weff,
                    const unsigned short* __restrict__ rwb,
                    const unsigned short* __restrict__ swb,
                    const float* __restrict__ biasW,
                    const float* __restrict__ rb,
                    const float* __restrict__ sb,
                    float* __restrict__ out_res,
                    float* __restrict__ out_skip) {
  extern __shared__ char smem[];
  const int XP = 72;                  // bf16 row pitch (pad vs bank conflicts)
  const int OP = S_ + 1;              // f32 row pitch for Out (257)
  float*          Xraw = (float*)smem;                     // [64][XRP] f32
  float*          Out  = (float*)smem;                     // [64][257] f32 (alias)
  unsigned short* Xa   = (unsigned short*)(smem + 66560);  // [256][72] bf16
  unsigned short* Yl   = (unsigned short*)(smem + 103424); // [256][72] bf16
  __shared__ float sBias[COUT], sRb[COUT], sSb[COUT];

  const int tid = threadIdx.x;
  const int blk = blockIdx.x;         // b*K + k
  const int b = blk >> 7, k = blk & 127;
  const int wave = tid >> 5, lane = tid & 31, hlf = lane >> 4;

  if (tid < COUT) {
    sBias[tid] = biasW[(size_t)blk * COUT + tid];
    sRb[tid]   = rb[tid];
    sSb[tid]   = sb[tid];
  }
  const float* xseg = x + (size_t)b * CIN * T_ + (size_t)k * S_;

#if HAS_ASYNC_LDS
  // 64 rows x 256 f32 = 64 KB, as 16 rounds of 256 lanes x 16B async copies.
  // Consecutive lanes cover consecutive 16B chunks of one x row -> each
  // instruction streams 512B contiguous from global into LDS, no VGPR data.
#pragma unroll
  for (int it = 0; it < 16; ++it) {
    int c = it * 256 + tid;           // 16-byte chunk id (64 rows x 64 chunks)
    int cin = c >> 6, tc = c & 63;
    const float* g = xseg + (size_t)cin * T_ + tc * 4;
    float* l = Xraw + cin * XRP + tc * 4;
    __builtin_amdgcn_global_load_async_to_lds_b128(
        (__attribute__((address_space(1))) v4i_*)g,
        (__attribute__((address_space(3))) v4i_*)l, 0, 0);
  }
  __builtin_amdgcn_s_wait_asynccnt(0);
  __syncthreads();
  // LDS->LDS transpose + f32->bf16 convert into the WMMA A layout.
  for (int i = tid; i < CIN * S_; i += 256) {
    int cin = i >> 8, tl = i & 255;
    Xa[tl * XP + cin] = f2bf(Xraw[cin * XRP + tl]);
  }
#else
  for (int i = tid; i < CIN * S_; i += 256) {
    int cin = i >> 8, tl = i & 255;
    Xa[tl * XP + cin] = f2bf(xseg[(size_t)cin * T_ + tl]);
  }
#endif
  __syncthreads();

  const unsigned short* wseg = weff + (size_t)blk * WROWS;  // [out][cin] bf16

  // ---- stage 1: Y = sin(X @ Weff + bias) ----
  for (int mi = 0; mi < 2; ++mi) {
    const int mt = wave * 2 + mi;
    for (int nt = 0; nt < 4; ++nt) {
      v8f acc = {};
      for (int kk = 0; kk < CIN; kk += 32) {
        v16bf a  = gatherA(Xa, XP, mt * 16, kk);
        v16bf bm = gatherB(wseg, CIN, nt * 16, kk);
        acc = __builtin_amdgcn_wmma_f32_16x16x32_bf16(false, a, false, bm,
                                                      (short)0, acc, false, false);
      }
      const int n = nt * 16 + (lane & 15);
      const float bias = sBias[n];
#pragma unroll
      for (int r = 0; r < 8; ++r) {
        int m = mt * 16 + r + 8 * hlf;
        Yl[m * XP + n] = f2bf(__sinf(acc[r] + bias));
      }
    }
  }
  __syncthreads();   // Xraw/Xa dead beyond here; Out may reuse region 0

  // ---- stage 2, pass R: Out = Y @ rw^T + rb ----
  for (int mi = 0; mi < 2; ++mi) {
    const int mt = wave * 2 + mi;
    for (int nt = 0; nt < 4; ++nt) {
      v8f acc = {};
      for (int kk = 0; kk < COUT; kk += 32) {
        v16bf a  = gatherA(Yl, XP, mt * 16, kk);
        v16bf bm = gatherB(rwb, COUT, nt * 16, kk);
        acc = __builtin_amdgcn_wmma_f32_16x16x32_bf16(false, a, false, bm,
                                                      (short)0, acc, false, false);
      }
      const int n = nt * 16 + (lane & 15);
      const float rbv = sRb[n];
#pragma unroll
      for (int r = 0; r < 8; ++r) {
        int m = mt * 16 + r + 8 * hlf;
        Out[n * OP + m] = acc[r] + rbv;
      }
    }
  }
  __syncthreads();
  // coalesced writer: out1 = (R + x) / 2   (x re-read is L2-hot)
  {
    float* ro = out_res + (size_t)b * COUT * T_ + (size_t)k * S_;
    for (int i = tid; i < COUT * S_; i += 256) {
      int o = i >> 8, tl = i & 255;
      float xv = xseg[(size_t)o * T_ + tl];
      ro[(size_t)o * T_ + tl] = (Out[o * OP + tl] + xv) * 0.5f;
    }
  }
  __syncthreads();

  // ---- stage 2, pass S: Out = Y @ sw^T + sb ----
  for (int mi = 0; mi < 2; ++mi) {
    const int mt = wave * 2 + mi;
    for (int nt = 0; nt < 4; ++nt) {
      v8f acc = {};
      for (int kk = 0; kk < COUT; kk += 32) {
        v16bf a  = gatherA(Yl, XP, mt * 16, kk);
        v16bf bm = gatherB(swb, COUT, nt * 16, kk);
        acc = __builtin_amdgcn_wmma_f32_16x16x32_bf16(false, a, false, bm,
                                                      (short)0, acc, false, false);
      }
      const int n = nt * 16 + (lane & 15);
      const float sbv = sSb[n];
#pragma unroll
      for (int r = 0; r < 8; ++r) {
        int m = mt * 16 + r + 8 * hlf;
        Out[n * OP + m] = acc[r] + sbv;
      }
    }
  }
  __syncthreads();
  {
    float* so = out_skip + (size_t)b * COUT * T_ + (size_t)k * S_;
    for (int i = tid; i < COUT * S_; i += 256) {
      int o = i >> 8, tl = i & 255;
      so[(size_t)o * T_ + tl] = Out[o * OP + tl];
    }
  }
}

// ---------------------------------------------------------------------------
extern "C" void kernel_launch(void* const* d_in, const int* in_sizes, int n_in,
                              void* d_out, int out_size, void* d_ws, size_t ws_size,
                              hipStream_t stream) {
  (void)in_sizes; (void)n_in; (void)out_size; (void)ws_size;
  const float* x   = (const float*)d_in[0];
  const float* z   = (const float*)d_in[1];
  const float* wm1 = (const float*)d_in[2];
  const float* wb1 = (const float*)d_in[3];
  const float* wm2 = (const float*)d_in[4];
  const float* wb2 = (const float*)d_in[5];
  const float* bm1 = (const float*)d_in[6];
  const float* bb1 = (const float*)d_in[7];
  const float* bm2 = (const float*)d_in[8];
  const float* bb2 = (const float*)d_in[9];
  const float* rw  = (const float*)d_in[10];
  const float* rb  = (const float*)d_in[11];
  const float* sw  = (const float*)d_in[12];
  const float* sb  = (const float*)d_in[13];

  char* ws = (char*)d_ws;
  unsigned short* hT    = (unsigned short*)(ws + 0);        // 131072 B
  unsigned short* w2e   = (unsigned short*)(ws + 131072);   // 262144 B
  float*          wb2e  = (float*)         (ws + 393216);   // 16384 B
  unsigned short* rwb   = (unsigned short*)(ws + 409600);   // 8192 B
  unsigned short* swb   = (unsigned short*)(ws + 417792);   // 8192 B
  float*          biasW = (float*)         (ws + 425984);   // 524288 B
  unsigned short* weff  = (unsigned short*)(ws + 950272);   // 16777216 B

  float* out_res  = (float*)d_out;
  float* out_skip = out_res + (size_t)B_ * COUT * T_;

  prep_kernel<<<320, 256, 0, stream>>>(z, wm1, wb1, wm2, wb2, bm1, bb1, bm2,
                                       bb2, rw, sw, hT, w2e, wb2e, rwb, swb,
                                       biasW);
  wgen_kernel<<<4096, 256, 0, stream>>>(hT, w2e, wb2e, weff);
  const int dynLds = 140288;          // Xraw/Out union + Xa + Y (fixed layout)
  hyperconv_main<<<B_ * K_, 256, dynLds, stream>>>(x, weff, rwb, swb, biasW,
                                                   rb, sb, out_res, out_skip);
}
